// MultiHeadAttention_81475529605855
// MI455X (gfx1250) — matrix operand order repaired and checked by reading, not compiled, not run on hardware
//
#include <hip/hip_runtime.h>

#define HIDDEN 1024
#define SEQ    1024
#define BATCH  8
#define HEADS  16
#define HDIM   64

typedef __attribute__((ext_vector_type(16))) __bf16 v16bf;
typedef __attribute__((ext_vector_type(8)))  float  v8f;

// One WMMA operand fragment: 8 VGPRs = 16 bf16 per lane.
union Frag {
  unsigned int   w[8];
  unsigned short u[16];
  v16bf          b;
};

// ---- CDNA5 async global->LDS copy (ASYNCcnt path), 16B per lane per op ----
// INST_OFFSET is added to BOTH the LDS and the global address (ISA 08 §4.4).
#define ASYNC_CP16(ldsaddr, gptr, OFFSTR)                                      \
  asm volatile("global_load_async_to_lds_b128 %0, %1, off offset:" OFFSTR     \
               :: "v"(ldsaddr), "v"(gptr) : "memory")
#define WAIT_ASYNC(NSTR) asm volatile("s_wait_asynccnt " NSTR ::: "memory")
#define WAIT_DS0()       asm volatile("s_wait_dscnt 0" ::: "memory")

__device__ __forceinline__ unsigned lds_addr32(const void* p) {
  return (unsigned)(unsigned long long)p;   // flat LDS aperture: addr[31:0] = DS offset
}

__device__ __forceinline__ unsigned short f2bf(float f) {
  unsigned int x = __float_as_uint(f);
  x += 0x7FFFu + ((x >> 16) & 1u);          // round-to-nearest-even
  return (unsigned short)(x >> 16);
}
__device__ __forceinline__ unsigned pack2(float a, float b) {
  return (unsigned)f2bf(a) | ((unsigned)f2bf(b) << 16);
}

// A-matrix fragment, contiguous K order (matches B): pair p covers K = kpair(p,hi), +1
__device__ __forceinline__ int kpair(int p, int hi) {
  return p * 2 + hi * 16;
}

__device__ __forceinline__ v8f wmma_bf16(const Frag& a, const Frag& b, v8f c) {
  return __builtin_amdgcn_wmma_f32_16x16x32_bf16(false, a.b, false, b.b,
                                                 (short)0, c, false, false);
}

// reductions across the 16 lanes sharing one output row (xor masks <16 stay in-group)
__device__ __forceinline__ float red_max16(float t) {
  t = fmaxf(t, __uint_as_float((unsigned)__builtin_amdgcn_ds_swizzle((int)__float_as_uint(t), 0x041F)));
  t = fmaxf(t, __uint_as_float((unsigned)__builtin_amdgcn_ds_swizzle((int)__float_as_uint(t), 0x081F)));
  t = fmaxf(t, __uint_as_float((unsigned)__builtin_amdgcn_ds_swizzle((int)__float_as_uint(t), 0x101F)));
  t = fmaxf(t, __uint_as_float((unsigned)__builtin_amdgcn_ds_swizzle((int)__float_as_uint(t), 0x201F)));
  return t;
}
__device__ __forceinline__ float red_sum16(float t) {
  t += __uint_as_float((unsigned)__builtin_amdgcn_ds_swizzle((int)__float_as_uint(t), 0x041F));
  t += __uint_as_float((unsigned)__builtin_amdgcn_ds_swizzle((int)__float_as_uint(t), 0x081F));
  t += __uint_as_float((unsigned)__builtin_amdgcn_ds_swizzle((int)__float_as_uint(t), 0x101F));
  t += __uint_as_float((unsigned)__builtin_amdgcn_ds_swizzle((int)__float_as_uint(t), 0x201F));
  return t;
}

__device__ __forceinline__ void loadA16(const float* __restrict__ src, unsigned short* dst) {
#pragma unroll
  for (int j = 0; j < 4; ++j) {
    float4 v = ((const float4*)src)[j];
    ((unsigned*)dst)[j * 2 + 0] = pack2(v.x, v.y);
    ((unsigned*)dst)[j * 2 + 1] = pack2(v.z, v.w);
  }
}
__device__ __forceinline__ void loadA16(const unsigned short* __restrict__ src, unsigned short* dst) {
#pragma unroll
  for (int j = 0; j < 2; ++j)
    ((uint4*)dst)[j] = ((const uint4*)src)[j];
}

__device__ __forceinline__ void storeC(float* p, float v) { *p = v; }
__device__ __forceinline__ void storeC(unsigned short* p, float v) { *p = f2bf(v); }

// Streaming fp32 -> bf16 convert of the 4 weight matrices, kept in [N][K] order
// (that IS W^T's B-matrix source layout for x @ W.T). Fully coalesced both sides.
__global__ __launch_bounds__(256) void convert_weights_kernel(
    const float* __restrict__ Wq, const float* __restrict__ Wk,
    const float* __restrict__ Wv, const float* __restrict__ Wo,
    unsigned short* __restrict__ out) {
  unsigned idx = blockIdx.x * 256u + threadIdx.x;     // 1M threads x 4 elems
  unsigned w   = idx >> 18;
  unsigned rem = (idx & 0x3FFFFu) * 4u;
  const float* W = (w == 0) ? Wq : (w == 1) ? Wk : (w == 2) ? Wv : Wo;
  float4 v = *(const float4*)(W + rem);
  uint2 o;
  o.x = pack2(v.x, v.y);
  o.y = pack2(v.z, v.w);
  *(uint2*)(out + ((size_t)w << 20) + rem) = o;
}

// C = cvt(A[M,K]) @ W^T + bias.  Bw is bf16 [N][K] (row n holds K contiguous).
// WG tile 128x128, 8 waves (4m x 2n), wave tile 32x64 -> 8 WMMA / K-step.
// TRANSV: write output transposed as [b, col, s] with one b128 store per tile.
template <typename InT, typename OutT, bool TRANSV>
__global__ __launch_bounds__(256) void gemm_bias_kernel(
    const InT* __restrict__ A, const unsigned short* __restrict__ Bw,
    const float* __restrict__ bias, OutT* __restrict__ C,
    int M, int N, int K, float outscale) {
  __shared__ unsigned short As[128 * 32];   // [m][k]
  __shared__ unsigned short Bs[128 * 32];   // [n][k]  (contiguous B-frag reads)

  const int tid  = threadIdx.x;
  const int wave = tid >> 5;
  const int lane = tid & 31;
  const int lm = lane & 15, hi = lane >> 4;
  const int wm = wave & 3;          // m offset wm*32
  const int wn = wave >> 2;         // n offset wn*64
  const int m0 = blockIdx.y * 128;
  const int n0 = blockIdx.x * 128;

  const int frow = tid >> 1, fcol = (tid & 1) * 16;   // tile fill: 16 elems/thread

  v8f acc[2][4] = {};

#pragma unroll 1
  for (int k0 = 0; k0 < K; k0 += 32) {
    loadA16(A + (size_t)(m0 + frow) * K + k0 + fcol, &As[frow * 32 + fcol]);
    {
      const unsigned short* s = Bw + (size_t)(n0 + frow) * K + k0 + fcol;
      *(uint4*)&Bs[frow * 32 + fcol]     = *(const uint4*)s;
      *(uint4*)&Bs[frow * 32 + fcol + 8] = *(const uint4*)(s + 8);
    }
    __syncthreads();

    Frag af[2], bf[4];
#pragma unroll
    for (int i = 0; i < 2; ++i) {
      const unsigned short* src = &As[(wm * 32 + i * 16 + lm) * 32];
#pragma unroll
      for (int p = 0; p < 8; ++p)
        af[i].w[p] = *(const unsigned int*)(src + kpair(p, hi));
    }
#pragma unroll
    for (int j = 0; j < 4; ++j) {           // B layout: K = e + 16*hi, contiguous
      const unsigned short* src = &Bs[(wn * 64 + j * 16 + lm) * 32 + 16 * hi];
#pragma unroll
      for (int p = 0; p < 8; ++p)
        bf[j].w[p] = *(const unsigned int*)(src + p * 2);
    }
#pragma unroll
    for (int i = 0; i < 2; ++i)
#pragma unroll
      for (int j = 0; j < 4; ++j)
        acc[i][j] = wmma_bf16(af[i], bf[j], acc[i][j]);
    __syncthreads();
  }

#pragma unroll
  for (int j = 0; j < 4; ++j) {
    const int col = n0 + wn * 64 + j * 16 + lm;
    const float bv = bias[col];
#pragma unroll
    for (int i = 0; i < 2; ++i) {
      if constexpr (TRANSV) {
        // rows r..r+7 are consecutive tokens -> pack 8 bf16 into one b128 store
        const int row0 = m0 + wm * 32 + i * 16 + 8 * hi;
        const int bb = row0 >> 10, s = row0 & 1023;
        uint4 q;
        q.x = pack2((acc[i][j][0] + bv) * outscale, (acc[i][j][1] + bv) * outscale);
        q.y = pack2((acc[i][j][2] + bv) * outscale, (acc[i][j][3] + bv) * outscale);
        q.z = pack2((acc[i][j][4] + bv) * outscale, (acc[i][j][5] + bv) * outscale);
        q.w = pack2((acc[i][j][6] + bv) * outscale, (acc[i][j][7] + bv) * outscale);
        *(uint4*)(C + (((size_t)((bb << 10) + col)) << 10) + s) = q;
      } else {
#pragma unroll
        for (int r = 0; r < 8; ++r) {
          const int row = m0 + wm * 32 + i * 16 + r + 8 * hi;
          storeC(C + (size_t)row * N + col, (acc[i][j][r] + bv) * outscale);
        }
      }
    }
  }
}

// Flash attention. 1 wave = 16 queries; WG = 64 queries of one (b,h).
// K/V tiles are WG-shared, double-buffered, filled with async global->LDS.
// Vw is the transposed V workspace [b, h*64+d, s]; Q pre-scaled by 1/8.
__global__ __launch_bounds__(128) void attention_kernel(
    const unsigned short* __restrict__ Qb, const unsigned short* __restrict__ Kb,
    const unsigned short* __restrict__ Vw, unsigned short* __restrict__ Ob) {
  __shared__ unsigned short Kt[2][32 * 64];   // [key][d]
  __shared__ unsigned short Vt[2][64 * 32];   // [d][key]  (contiguous V-frag reads)
  __shared__ unsigned short PtS[4][16 * 32];

  const int tid  = threadIdx.x;
  const int wave = tid >> 5;
  const int lane = tid & 31;
  const int lm = lane & 15, hi = lane >> 4;
  const int b = blockIdx.z, h = blockIdx.y;
  const int q0 = blockIdx.x * 64 + wave * 16;
  unsigned short* pt = PtS[wave];

  const unsigned short* kbase = Kb + (size_t)(b * SEQ) * HIDDEN + h * HDIM;
  const unsigned short* vbase = Vw + (size_t)((b << 10) + h * HDIM) * SEQ;

  // cooperative async fill: K: thread -> (key=tid/4, 32B seg), V: (d=tid/2, 32B seg)
  const int krow = tid >> 2, kseg = (tid & 3) * 16;
  const int vrow = tid >> 1, vseg = (tid & 1) * 16;

  auto issue_tile = [&](int kb, int p) {
    unsigned kl = lds_addr32(&Kt[p][krow * 64 + kseg]);
    const unsigned short* kg = kbase + (size_t)(kb + krow) * HIDDEN + kseg;
    ASYNC_CP16(kl, kg, "0");
    ASYNC_CP16(kl, kg, "16");
    unsigned vl = lds_addr32(&Vt[p][vrow * 32 + vseg]);
    const unsigned short* vg = vbase + (size_t)vrow * SEQ + kb + vseg;
    ASYNC_CP16(vl, vg, "0");
    ASYNC_CP16(vl, vg, "16");
  };

  // persistent Q fragments (A layout), head dim 64 -> two 32-chunks
  const unsigned short* qrow = Qb + (size_t)(b * SEQ + q0 + lm) * HIDDEN + h * HDIM;
  Frag aq[2];
#pragma unroll
  for (int f = 0; f < 2; ++f)
#pragma unroll
    for (int p = 0; p < 8; ++p)
      aq[f].w[p] = *(const unsigned int*)(qrow + f * 32 + kpair(p, hi));

  float mrow[8], lrow[8];
  v8f oacc[4] = {};
#pragma unroll
  for (int r = 0; r < 8; ++r) { mrow[r] = -1e30f; lrow[r] = 0.f; }

  issue_tile(0, 0);

#pragma unroll 1
  for (int kb = 0; kb < SEQ; kb += 32) {
    const int p = (kb >> 5) & 1;
    WAIT_DS0();
    __syncthreads();                         // all waves done reading buffer p^1
    int nkb = kb + 32;
    if (nkb >= SEQ) nkb = 0;                 // harmless wrap prefetch on last iter
    issue_tile(nkb, p ^ 1);
    WAIT_ASYNC("4");                         // my 4 older ops (tile p) landed
    __syncthreads();                         // -> whole tile p valid
    const unsigned short* kt = Kt[p];
    const unsigned short* vt = Vt[p];

    // scores S = Q @ K^T : 2 key n-tiles x 2 K-chunks
    v8f sc[2];
#pragma unroll
    for (int j = 0; j < 2; ++j) {
      v8f s = {};
#pragma unroll
      for (int f = 0; f < 2; ++f) {
        Frag bkf;
        const unsigned short* src = kt + (j * 16 + lm) * 64 + f * 32 + hi * 16;
#pragma unroll
        for (int pp = 0; pp < 8; ++pp)
          bkf.w[pp] = *(const unsigned int*)(src + pp * 2);
        s = wmma_bf16(aq[f], bkf, s);
      }
      sc[j] = s;
    }

    // online softmax (row = r + 8*hi, replicated across each 16-lane group)
    float alpha[8];
#pragma unroll
    for (int r = 0; r < 8; ++r) {
      float t = red_max16(fmaxf(sc[0][r], sc[1][r]));
      float mn = fmaxf(mrow[r], t);
      alpha[r] = __expf(mrow[r] - mn);
      mrow[r] = mn;
    }
#pragma unroll
    for (int r = 0; r < 8; ++r) {
      float p0 = __expf(sc[0][r] - mrow[r]);
      float p1 = __expf(sc[1][r] - mrow[r]);
      pt[(r + 8 * hi) * 32 + lm]      = f2bf(p0);
      pt[(r + 8 * hi) * 32 + 16 + lm] = f2bf(p1);
      lrow[r] = lrow[r] * alpha[r] + red_sum16(p0 + p1);
    }
#pragma unroll
    for (int d = 0; d < 4; ++d)
#pragma unroll
      for (int r = 0; r < 8; ++r) oacc[d][r] *= alpha[r];

    // O += P @ V : P back through LDS in A layout; V fragments contiguous
    Frag ap;
#pragma unroll
    for (int pp = 0; pp < 8; ++pp)
      ap.w[pp] = *(const unsigned int*)(pt + lm * 32 + kpair(pp, hi));
#pragma unroll
    for (int d = 0; d < 4; ++d) {
      Frag bvf;
      const unsigned short* src = vt + (d * 16 + lm) * 32 + 16 * hi;
#pragma unroll
      for (int pp = 0; pp < 8; ++pp)
        bvf.w[pp] = *(const unsigned int*)(src + pp * 2);
      oacc[d] = wmma_bf16(ap, bvf, oacc[d]);
    }
  }

  // normalize & store bf16 O row-major [B*S, H*HDIM] (A-input of final GEMM)
#pragma unroll
  for (int d = 0; d < 4; ++d)
#pragma unroll
    for (int r = 0; r < 8; ++r) {
      float v = oacc[d][r] / lrow[r];
      Ob[(size_t)(b * SEQ + q0 + r + 8 * hi) * HIDDEN + h * HDIM + d * 16 + lm] = f2bf(v);
    }
}

extern "C" void kernel_launch(void* const* d_in, const int* in_sizes, int n_in,
                              void* d_out, int out_size, void* d_ws, size_t ws_size,
                              hipStream_t stream) {
  (void)in_sizes; (void)n_in; (void)out_size; (void)ws_size;
  const float* query = (const float*)d_in[0];
  const float* keyi  = (const float*)d_in[1];
  const float* value = (const float*)d_in[2];
  const float* Wq = (const float*)d_in[3];
  const float* bq = (const float*)d_in[4];
  const float* Wk = (const float*)d_in[5];
  const float* bk = (const float*)d_in[6];
  const float* Wv = (const float*)d_in[7];
  const float* bv = (const float*)d_in[8];
  const float* Wo = (const float*)d_in[9];
  const float* bo = (const float*)d_in[10];

  char* ws = (char*)d_ws;
  unsigned short* Wbf = (unsigned short*)ws;                 // 4 x 1Mi bf16 (8 MB)
  unsigned short* WqB = Wbf + 0ull * (1u << 20);
  unsigned short* WkB = Wbf + 1ull * (1u << 20);
  unsigned short* WvB = Wbf + 2ull * (1u << 20);
  unsigned short* WoB = Wbf + 3ull * (1u << 20);
  unsigned short* Qb  = (unsigned short*)(ws + (8ull  << 20));  // 16 MB each
  unsigned short* Kbf = (unsigned short*)(ws + (24ull << 20));
  unsigned short* Vtr = (unsigned short*)(ws + (40ull << 20));  // transposed V
  unsigned short* Obf = (unsigned short*)(ws + (56ull << 20));

  convert_weights_kernel<<<dim3(4096), dim3(256), 0, stream>>>(Wq, Wk, Wv, Wo, Wbf);

  const int M = BATCH * SEQ;
  dim3 ggrid(HIDDEN / 128, M / 128);
  // fold 1/sqrt(HEAD_DIM) = 1/8 into Q (bias included -> exact)
  gemm_bias_kernel<float, unsigned short, false><<<ggrid, 256, 0, stream>>>(
      query, WqB, bq, Qb, M, HIDDEN, HIDDEN, 0.125f);
  gemm_bias_kernel<float, unsigned short, false><<<ggrid, 256, 0, stream>>>(
      keyi, WkB, bk, Kbf, M, HIDDEN, HIDDEN, 1.0f);
  gemm_bias_kernel<float, unsigned short, true><<<ggrid, 256, 0, stream>>>(
      value, WvB, bv, Vtr, M, HIDDEN, HIDDEN, 1.0f);

  attention_kernel<<<dim3(SEQ / 64, HEADS, BATCH), 128, 0, stream>>>(Qb, Kbf, Vtr, Obf);

  gemm_bias_kernel<unsigned short, float, false><<<ggrid, 256, 0, stream>>>(
      Obf, WoB, bo, (float*)d_out, M, HIDDEN, HIDDEN, 1.0f);
}
